// EncoderLayer_41798621725071
// MI455X (gfx1250) — matrix operand order repaired and checked by reading, not compiled, not run on hardware
//
#include <hip/hip_runtime.h>

typedef __bf16 bf16;
typedef __attribute__((ext_vector_type(16))) __bf16 v16bf;
typedef __attribute__((ext_vector_type(8)))  __bf16 bf16x8;
typedef __attribute__((ext_vector_type(8)))  float   v8f;

#define S_LEN  4096
#define H_DIM  256
#define NHEAD  8
#define DHEAD  32
#define WWIN   128
#define FF_DIM 1024
#define MROWS  16384   // B*S
#define NEGBIG (-2.0e9f)

__device__ __forceinline__ float gelu_f(float x) {
  return 0.5f * x * (1.0f + erff(x * 0.70710678118654752440f));
}

__device__ __forceinline__ v16bf zero16() {
  v16bf z;
#pragma unroll
  for (int i = 0; i < 16; ++i) z[i] = (bf16)0.0f;
  return z;
}

// A-fragment: p = row_base + hi*16; lane<16 holds K {0..15}, lane>=16 {16..31}
__device__ __forceinline__ v16bf load_a(const bf16* p) {
  union { v16bf v; bf16x8 h[2]; } u;
  u.h[0] = *(const bf16x8*)(p);
  u.h[1] = *(const bf16x8*)(p + 8);
  return u.v;
}
// B-fragment: p = row_base + hi*16; contiguous 16 K-values of column N=lane&15
__device__ __forceinline__ v16bf load_b(const bf16* p) {
  union { v16bf v; bf16x8 h[2]; } u;
  u.h[0] = *(const bf16x8*)(p);
  u.h[1] = *(const bf16x8*)(p + 8);
  return u.v;
}

__device__ __forceinline__ v8f wmma_bf16(v16bf a, v16bf b, v8f c) {
  return __builtin_amdgcn_wmma_f32_16x16x32_bf16(false, a, false, b, (short)0, c, false, false);
}

// ---------------- LayerNorm (ddof=1, eps added to std) ----------------
template <typename OUT>
__global__ __launch_bounds__(128) void ln_kernel(const float* __restrict__ x,
                                                 const float* __restrict__ g,
                                                 const float* __restrict__ bb,
                                                 OUT* __restrict__ out) {
  const int row  = blockIdx.x * 4 + (threadIdx.x >> 5);
  const int lane = threadIdx.x & 31;
  const float* xr = x + (size_t)row * H_DIM;
  float v[8];
  float s = 0.f;
#pragma unroll
  for (int i = 0; i < 8; ++i) { v[i] = xr[i * 32 + lane]; s += v[i]; }
#pragma unroll
  for (int o = 16; o > 0; o >>= 1) s += __shfl_xor(s, o);
  const float mean = s * (1.0f / 256.0f);
  float q = 0.f;
#pragma unroll
  for (int i = 0; i < 8; ++i) { float d = v[i] - mean; q += d * d; }
#pragma unroll
  for (int o = 16; o > 0; o >>= 1) q += __shfl_xor(q, o);
  const float denom = sqrtf(q * (1.0f / 255.0f)) + 1e-9f;
  OUT* orow = out + (size_t)row * H_DIM;
#pragma unroll
  for (int i = 0; i < 8; ++i) {
    int c = i * 32 + lane;
    orow[c] = (OUT)(g[c] * (v[i] - mean) / denom + bb[c]);
  }
}

// --------- depthwise 9-tap conv (scalar taps) + gelu + 3x pointwise gelu ---------
__global__ __launch_bounds__(256) void dwpw_kernel(const float* __restrict__ xn,
    const float* __restrict__ dw_w, const float* __restrict__ dw_b,
    const float* __restrict__ pw_w, const float* __restrict__ pw_b,
    bf16* __restrict__ a0, bf16* __restrict__ a1, bf16* __restrict__ a2) {
  const int id = blockIdx.x * 256 + threadIdx.x;
  const int h  = id & (H_DIM - 1);
  const int bs = id >> 8;
  const int s  = bs & (S_LEN - 1);
  const int bbase = bs - s;
  float acc = 0.f;
#pragma unroll
  for (int t = 0; t < 9; ++t) {
    int sp = s + t - 4;
    if (sp >= 0 && sp < S_LEN)
      acc += dw_w[t] * xn[((size_t)(bbase + sp)) * H_DIM + h];
  }
  acc += dw_b[0];
  const float hm = gelu_f(acc);
  a0[id] = (bf16)gelu_f(hm * pw_w[0] + pw_b[0]);
  a1[id] = (bf16)gelu_f(hm * pw_w[1] + pw_b[1]);
  a2[id] = (bf16)gelu_f(hm * pw_w[2] + pw_b[2]);
}

// ---------------- weight convert + transpose: src[K,N] f32 -> dst[N,K] bf16 ----------------
__global__ __launch_bounds__(256) void transpose_bf16_kernel(const float* __restrict__ src,
    bf16* __restrict__ dst, int K, int N) {
  const int id = blockIdx.x * 256 + threadIdx.x;
  if (id >= K * N) return;
  const int k = id / N, n = id - k * N;
  dst[(size_t)n * K + k] = (bf16)src[id];
}

// ---------------- generic WMMA GEMM with conv-tap A addressing ----------------
// B tile (128 cols x 32 k, shared by all 8 waves) is staged in LDS with
// double-buffered global_load_async_to_lds_b128 (ASYNCcnt) so the DMA of
// tile kb+1 overlaps the WMMAs of tile kb.
// EPI: 0 = qkv scatter (bf16, [B,NH,S,DH], *scale)
//      1 = outF = acc + resid            (fp32)
//      2 = outB = bf16(gelu(acc + bias))
//      3 = outF = resid + gelu(acc + bias)
template <int TAPS, int DIL, int KPER, int NCOLS, int EPI>
__global__ __launch_bounds__(256) void gemm_kernel(
    const bf16* __restrict__ A, const bf16* __restrict__ WT,
    const float* __restrict__ bias, const float* __restrict__ resid,
    float* __restrict__ outF, bf16* __restrict__ outB, float scale) {
  __shared__ bf16 Bt[2][128][40];   // 32 valid k + 8 pad (80B row stride, conflict-free)
  const int lane = threadIdx.x & 31;
  const int w    = threadIdx.x >> 5;
  const int wm = w & 3, wn = w >> 2;
  const int m0 = blockIdx.y * 64 + wm * 16;
  const int n0_blk = blockIdx.x * 128;
  const int n0 = n0_blk + wn * 64;
  const int l15 = lane & 15, hi = lane >> 4;
  const int K = TAPS * KPER;
  const int mrow = m0 + l15;
  const int s = mrow & (S_LEN - 1);
  const int bbase = mrow - s;

  // async B-tile stage: each wave moves its 1KB share (2 x 32 lanes x 16B)
  const int st_row   = w * 16 + (lane >> 2);   // +8 on second issue
  const int st_chunk = (lane & 3) * 16;        // byte chunk within 64B row
  auto issueB = [&](int buf, int kb2) {
#pragma unroll
    for (int i = 0; i < 2; ++i) {
      const int r2 = st_row + i * 8;
      const unsigned voff = (unsigned)(n0_blk + r2) * (unsigned)(K * 2) +
                            (unsigned)(kb2 * 64 + st_chunk);
      const unsigned lds = (unsigned)(uintptr_t)(&Bt[buf][r2][0]) + (unsigned)st_chunk;
      asm volatile("global_load_async_to_lds_b128 %0, %1, %2"
                   :: "v"(lds), "v"(voff), "s"(WT) : "memory");
    }
  };

  const v8f z8 = {0.f,0.f,0.f,0.f,0.f,0.f,0.f,0.f};
  v8f acc[4];
#pragma unroll
  for (int nt = 0; nt < 4; ++nt) acc[nt] = z8;

  const int KB = K / 32;
  issueB(0, 0);
  for (int kb = 0; kb < KB; ++kb) {
    const int cur = kb & 1;
    asm volatile("s_wait_asynccnt 0x0" ::: "memory");  // own async portion landed
    __syncthreads();                                   // whole tile published
    if (kb + 1 < KB) issueB(cur ^ 1, kb + 1);

    const int t  = kb / (KPER / 32);
    const int kk = (kb - t * (KPER / 32)) * 32;
    const int se = s + (t - (TAPS - 1)) * DIL;
    v16bf a;
    if (se >= 0) {
      const bf16* p = A + ((size_t)(bbase + se)) * KPER + kk + hi * 16;
      a = load_a(p);
    } else {
      a = zero16();
    }
#pragma unroll
    for (int nt = 0; nt < 4; ++nt) {
      const int nl = wn * 64 + nt * 16 + l15;          // column within block tile
      v16bf bmat = load_b(&Bt[cur][nl][hi * 16]);
      acc[nt] = wmma_bf16(a, bmat, acc[nt]);
    }
  }

  const int Mof = hi * 8;
#pragma unroll
  for (int nt = 0; nt < 4; ++nt) {
#pragma unroll
    for (int r = 0; r < 8; ++r) {
      const int m    = m0 + r + Mof;
      const int ncol = n0 + nt * 16 + l15;
      float val = acc[nt][r];
      if (EPI == 0) {
        const int nh = ncol >> 5, dh = ncol & 31;
        const int b = m >> 12, ss = m & (S_LEN - 1);
        outB[(((size_t)(b * NHEAD + nh)) * S_LEN + ss) * DHEAD + dh] = (bf16)(val * scale);
      } else if (EPI == 1) {
        const size_t idx = (size_t)m * NCOLS + ncol;
        outF[idx] = val + resid[idx];
      } else if (EPI == 2) {
        outB[(size_t)m * NCOLS + ncol] = (bf16)gelu_f(val + bias[ncol]);
      } else {
        const size_t idx = (size_t)m * NCOLS + ncol;
        outF[idx] = resid[idx] + gelu_f(val + bias[ncol]);
      }
    }
  }
}

// ---------------- sliding-window attention: 1 wave per 16-query tile ----------------
__global__ __launch_bounds__(32) void attn_kernel(
    const bf16* __restrict__ q, const bf16* __restrict__ k, const bf16* __restrict__ v,
    const float* __restrict__ rel_bias, bf16* __restrict__ ctx) {
  __shared__ float sc[16][400];    // fp32 scores/exp  (25600 B)
  __shared__ bf16  pr[16][400];    // bf16 probs       (12800 B)
  __shared__ bf16  vT[32][400];    // v transposed     (25600 B)
  __shared__ float bias_s[260];    // rel_bias[0..256] staged in LDS
  const int mt = blockIdx.x, hh = blockIdx.y, b = blockIdx.z;
  const int lane = threadIdx.x;
  const int l15 = lane & 15, hi = lane >> 4;
  const int i0 = mt * 16;
  const int j0 = (i0 / WWIN) * WWIN - WWIN;   // window start (384 keys)
  const size_t headoff = ((size_t)(b * NHEAD + hh)) * S_LEN * DHEAD;
  const bf16* qh = q + headoff;
  const bf16* kh = k + headoff;
  const bf16* vh = v + headoff;

  for (int idx = lane; idx < 257; idx += 32) bias_s[idx] = rel_bias[idx];

  v16bf qa = load_a(qh + (size_t)(i0 + l15) * DHEAD + hi * 16);

  // stage v window transposed into LDS (zero-padded outside [0,S))
  for (int it = 0; it < 12; ++it) {
    const int kk = it * 32 + lane;
    const int j = j0 + kk;
    if (j >= 0 && j < S_LEN) {
      const bf16* p = vh + (size_t)j * DHEAD;
#pragma unroll
      for (int d = 0; d < 32; ++d) vT[d][kk] = p[d];
    } else {
#pragma unroll
      for (int d = 0; d < 32; ++d) vT[d][kk] = (bf16)0.0f;
    }
  }
  __syncthreads();

  const v8f z8 = {0.f,0.f,0.f,0.f,0.f,0.f,0.f,0.f};
  // scores: 24 key tiles, one WMMA each (K = DH = 32)
  for (int nt = 0; nt < 24; ++nt) {
    const int kk = nt * 16 + l15;
    const int j = j0 + kk;
    const bool jvalid = (j >= 0 && j < S_LEN);
    v16bf kbf;
    if (jvalid) kbf = load_b(kh + (size_t)j * DHEAD + hi * 16);
    else        kbf = zero16();
    v8f sv = wmma_bf16(qa, kbf, z8);
#pragma unroll
    for (int r = 0; r < 8; ++r) {
      const int M = r + hi * 8;
      const int rel = j - (i0 + M);
      const int ridx = min(max(rel + WWIN, 0), 2 * WWIN);
      const bool valid = jvalid && (rel >= -WWIN) && (rel <= WWIN);
      // additive penalty keeps the LDS bias load unconditional (no exec-mask
      // branches); invalid slots end up ~ -2e9 -> exp underflows to exactly 0
      const float pen = valid ? 0.0f : NEGBIG;
      sc[M][kk] = sv[r] + bias_s[ridx] + pen;
    }
  }
  __syncthreads();

  // fp32 softmax: lane pair (l15, l15+16) shares a row, halves of 384 cols
  {
    const int row = l15, c0 = hi * 192;
    float m = -3.4e38f;
    for (int x = 0; x < 192; ++x) m = fmaxf(m, sc[row][c0 + x]);
    m = fmaxf(m, __shfl_xor(m, 16));
    float ssum = 0.f;
    for (int x = 0; x < 192; ++x) {
      float e = __expf(sc[row][c0 + x] - m);
      sc[row][c0 + x] = e;
      ssum += e;
    }
    ssum += __shfl_xor(ssum, 16);
    const float inv = 1.0f / ssum;
    for (int x = 0; x < 192; ++x) pr[row][c0 + x] = (bf16)(sc[row][c0 + x] * inv);
  }
  __syncthreads();

  // P @ V : 12 K-tiles of 32, 2 N-tiles (DH=32)
  v8f acc0 = z8, acc1 = z8;
  for (int kt = 0; kt < 12; ++kt) {
    v16bf pa = load_a(&pr[l15][kt * 32 + hi * 16]);
    v16bf b0 = load_b(&vT[l15][kt * 32 + hi * 16]);
    v16bf b1 = load_b(&vT[16 + l15][kt * 32 + hi * 16]);
    acc0 = wmma_bf16(pa, b0, acc0);
    acc1 = wmma_bf16(pa, b1, acc1);
  }

#pragma unroll
  for (int r = 0; r < 8; ++r) {
    const int iq = i0 + r + hi * 8;
    const size_t rowoff = ((size_t)(b * S_LEN + iq)) * H_DIM + hh * DHEAD;
    ctx[rowoff + l15]      = (bf16)acc0[r];
    ctx[rowoff + 16 + l15] = (bf16)acc1[r];
  }
}

// ============================== host side ==============================
extern "C" void kernel_launch(void* const* d_in, const int* in_sizes, int n_in,
                              void* d_out, int out_size, void* d_ws, size_t ws_size,
                              hipStream_t stream) {
  (void)in_sizes; (void)n_in; (void)out_size; (void)ws_size;
  const float* x_in  = (const float*)d_in[0];
  const float* dw_w  = (const float*)d_in[1];
  const float* dw_b  = (const float*)d_in[2];
  const float* pw_w  = (const float*)d_in[3];
  const float* pw_b  = (const float*)d_in[4];
  const float* Wq    = (const float*)d_in[5];
  const float* Wk    = (const float*)d_in[6];
  const float* Wv    = (const float*)d_in[7];
  const float* Wo    = (const float*)d_in[8];
  const float* rel_b = (const float*)d_in[9];
  const float* ln1g  = (const float*)d_in[10];
  const float* ln1b  = (const float*)d_in[11];
  const float* ln2g  = (const float*)d_in[12];
  const float* ln2b  = (const float*)d_in[13];
  const float* w1    = (const float*)d_in[14];
  const float* b1    = (const float*)d_in[15];
  const float* w2    = (const float*)d_in[16];
  const float* b2    = (const float*)d_in[17];
  float* out = (float*)d_out;

  char* ws = (char*)d_ws;
  const size_t SZ_F32 = (size_t)MROWS * H_DIM * 4;   // 16 MB
  const size_t SZ_BF  = (size_t)MROWS * H_DIM * 2;   //  8 MB
  float* xn_x2 = (float*)(ws);                       // LN1 out, later x2 (residual stream)
  bf16* act0 = (bf16*)(ws + SZ_F32);
  bf16* act1 = (bf16*)(ws + SZ_F32 + SZ_BF);
  bf16* act2 = (bf16*)(ws + SZ_F32 + 2 * SZ_BF);
  bf16* ctx  = act0;                                 // reuse after QKV GEMMs
  bf16* xn2  = act1;                                 // reuse after QKV GEMMs
  bf16* qb   = (bf16*)(ws + SZ_F32 + 3 * SZ_BF);
  bf16* kb   = (bf16*)(ws + SZ_F32 + 4 * SZ_BF);
  bf16* vb   = (bf16*)(ws + SZ_F32 + 5 * SZ_BF);
  bf16* fbuf = qb;                                   // 32 MB, reuse q/k/v region after attention
  char* wsw = ws + SZ_F32 + 3 * SZ_BF + (size_t)MROWS * FF_DIM * 2;
  bf16* WTq = (bf16*)(wsw);
  bf16* WTk = (bf16*)(wsw + 1u * 131072);
  bf16* WTv = (bf16*)(wsw + 2u * 131072);
  bf16* WTo = (bf16*)(wsw + 3u * 131072);
  bf16* WT1 = (bf16*)(wsw + 4u * 131072);
  bf16* WT2 = (bf16*)(wsw + 4u * 131072 + 4718592u);

  // weight convert + transpose (bf16, [N,K])
  transpose_bf16_kernel<<<(65536 + 255) / 256, 256, 0, stream>>>(Wq, WTq, 256, 256);
  transpose_bf16_kernel<<<(65536 + 255) / 256, 256, 0, stream>>>(Wk, WTk, 256, 256);
  transpose_bf16_kernel<<<(65536 + 255) / 256, 256, 0, stream>>>(Wv, WTv, 256, 256);
  transpose_bf16_kernel<<<(65536 + 255) / 256, 256, 0, stream>>>(Wo, WTo, 256, 256);
  transpose_bf16_kernel<<<(2359296 + 255) / 256, 256, 0, stream>>>(w1, WT1, 2304, 1024);
  transpose_bf16_kernel<<<(2359296 + 255) / 256, 256, 0, stream>>>(w2, WT2, 9216, 256);

  // LN1 -> xn (fp32)
  ln_kernel<float><<<MROWS / 4, 128, 0, stream>>>(x_in, ln1g, ln1b, xn_x2);
  // depthwise conv + gelu + pointwise gelu -> act0/1/2 (bf16)
  dwpw_kernel<<<(MROWS * H_DIM) / 256, 256, 0, stream>>>(xn_x2, dw_w, dw_b, pw_w, pw_b,
                                                         act0, act1, act2);
  // Q/K/V projections (q scaled by DH^-0.5), scatter to [B,NH,S,DH] bf16
  dim3 g256(256 / 128, MROWS / 64);
  gemm_kernel<1, 1, 256, 256, 0><<<g256, 256, 0, stream>>>(act0, WTq, nullptr, nullptr,
      nullptr, qb, 0.17677669529663687f);
  gemm_kernel<1, 1, 256, 256, 0><<<g256, 256, 0, stream>>>(act1, WTk, nullptr, nullptr,
      nullptr, kb, 1.0f);
  gemm_kernel<1, 1, 256, 256, 0><<<g256, 256, 0, stream>>>(act2, WTv, nullptr, nullptr,
      nullptr, vb, 1.0f);
  // sliding-window attention -> ctx (bf16, [B*S, H])
  attn_kernel<<<dim3(S_LEN / 16, NHEAD, 4), 32, 0, stream>>>(qb, kb, vb, rel_b, ctx);
  // x2 = x + ctx @ Wo   (fp32)
  gemm_kernel<1, 1, 256, 256, 1><<<g256, 256, 0, stream>>>(ctx, WTo, nullptr, x_in,
      xn_x2, nullptr, 1.0f);
  // LN2 -> xn2 (bf16)
  ln_kernel<bf16><<<MROWS / 4, 128, 0, stream>>>(xn_x2, ln2g, ln2b, xn2);
  // f = gelu(conv1(xn2)): taps t-8, dil 1, K = 9*256, N = 1024 -> bf16
  gemm_kernel<9, 1, 256, 1024, 2><<<dim3(1024 / 128, MROWS / 64), 256, 0, stream>>>(
      xn2, WT1, b1, nullptr, nullptr, fbuf, 1.0f);
  // out = x2 + gelu(conv2(f)): taps (t-8)*8, K = 9*1024, N = 256 -> fp32
  gemm_kernel<9, 8, 1024, 256, 3><<<g256, 256, 0, stream>>>(
      fbuf, WT2, b2, xn_x2, out, nullptr, 1.0f);
}
